// CAGLocal_34153579938464
// MI455X (gfx1250) — matrix operand order repaired and checked by reading, not compiled, not run on hardware
//
#include <hip/hip_runtime.h>
#include <hip/hip_bf16.h>

typedef __attribute__((ext_vector_type(16))) __bf16 v16bf;
typedef __attribute__((ext_vector_type(8)))  float  v8f;
typedef __attribute__((ext_vector_type(4)))  unsigned int u32x4;
typedef __attribute__((ext_vector_type(8)))  int i32x8;
typedef __attribute__((ext_vector_type(4)))  int i32x4;

#define C_DIM 64
#define H_DIM 128
#define W_DIM 128
#define KWIN  5
#define HKV   (H_DIM * KWIN)   // 640
#define WKV   (W_DIM * KWIN)   // 640
#define QT    16               // queries per workgroup (1 row x 16 cols)
#define PX    (QT * KWIN * KWIN) // 400 kv pixels per tile
#define NTILE (PX / 16)        // 25 N-tiles of 16 px
#define COLS  (QT * KWIN)      // 80 kv columns per tile

// LDS byte offsets (all 16B aligned)
#define XF_BYTES   (C_DIM * PX * 2)          // 51200  bf16 B-fragments of feat_kv tile
#define WF_BYTES   (2 * C_DIM * C_DIM * 2)   // 16384  bf16 A-fragments of Wk,Wv
#define KS_OFF     (XF_BYTES + WF_BYTES)     // 67584; fp32 TDM stage aliases Ks

__global__ __launch_bounds__(256, 1)
void cag_local_fused(const float* __restrict__ feat_q,
                     const float* __restrict__ feat_kv,
                     const float* __restrict__ Wq,
                     const float* __restrict__ Wk,
                     const float* __restrict__ Wv,
                     float* __restrict__ out)
{
    extern __shared__ __align__(16) char smem[];
    __bf16* Xf   = (__bf16*)smem;                 // [2 kstep][25 nt][32 lane][16] bf16
    __bf16* Wf   = (__bf16*)(smem + XF_BYTES);    // [2 mat][4 mt][2 kstep][32 lane][16]
    float*  Ks   = (float*)(smem + KS_OFF);       // [64 ch][400 px]; first used as fp32 stage
    float*  Vs   = Ks + C_DIM * PX;
    float*  qs   = Vs + C_DIM * PX;               // [16 l][64 c]  (pre-scaled by 1/8)
    float*  fqs  = qs + QT * C_DIM;               // [64 c][16 l]
    float*  attnB = fqs + C_DIM * QT;             // [16 l][25 n]

    const int tid = threadIdx.x;
    const int qxb = blockIdx.x;          // 0..7
    const int qy  = blockIdx.y;          // 0..127
    const int b   = blockIdx.z;          // 0..B-1
    const int qx0 = qxb * QT;

    const long kvplane = (long)HKV * WKV;       // 409600
    const long qplane  = (long)H_DIM * W_DIM;   // 16384
    const float* kvbase = feat_kv + (long)b * C_DIM * kvplane
                                  + (long)(qy * KWIN) * WKV + (long)(qx0 * KWIN);
    const float* qbase  = feat_q + (long)b * C_DIM * qplane
                                  + (long)qy * W_DIM + qx0;
    float* obase = out + (long)b * C_DIM * qplane + (long)qy * W_DIM + qx0;

    // ---- Kick off TDM: DMA the 64ch x 5row x 80col fp32 tile of feat_kv into LDS ----
    // Issued once (wave 0 only: TDM ignores EXEC, so guard at wave granularity).
    if (tid == 0) {
        unsigned long long gaddr = (unsigned long long)(uintptr_t)kvbase;
        unsigned int ldsa = (unsigned int)(uintptr_t)Ks;   // flat addr low 32 = LDS offset

        u32x4 g0;
        g0[0] = 1u;                                        // count=1, user descriptor
        g0[1] = ldsa;                                      // lds_addr (bytes)
        g0[2] = (unsigned int)(gaddr & 0xffffffffu);       // global_addr[31:0]
        g0[3] = (unsigned int)((gaddr >> 32) & 0x1ffffffu) // global_addr[56:32]
              | (2u << 30);                                // type=2 ("image")

        i32x8 g1;
        g1[0] = (int)(2u << 16);                           // wg_mask=0, data_size=4B
        g1[1] = (int)((unsigned)(WKV & 0xffff) << 16);     // tensor_dim0 lo16
        g1[2] = (int)((unsigned)(WKV >> 16) | ((unsigned)(HKV & 0xffff) << 16));
        g1[3] = (int)((unsigned)(HKV >> 16) | ((unsigned)COLS << 16));   // tile_dim0=80
        g1[4] = (int)((unsigned)KWIN | ((unsigned)C_DIM << 16));         // tile_dim1=5, tile_dim2=64
        g1[5] = (int)WKV;                                  // tensor_dim0_stride lo32 = 640
        g1[6] = (int)(((unsigned)(kvplane & 0xffff)) << 16);             // stride0 hi16=0 | stride1 lo16
        g1[7] = (int)((unsigned long long)kvplane >> 16);  // tensor_dim1_stride[47:16]

        i32x4 g2;
        g2[0] = C_DIM;                                     // tensor_dim2 = 64 channels
        g2[1] = 0;                                         // tensor_dim3 unused
        g2[2] = (int)kvplane;                              // tensor_dim2_stride lo32 = 409600
        g2[3] = 0;                                         // stride2 hi16 | tile_dim3=0

        i32x4 g3 = (i32x4){0, 0, 0, 0};

#if __clang_major__ >= 23
        i32x8 gz = (i32x8){0, 0, 0, 0, 0, 0, 0, 0};
        __builtin_amdgcn_tensor_load_to_lds(g0, g1, g2, g3, gz, 0);
#else
        __builtin_amdgcn_tensor_load_to_lds(g0, g1, g2, g3, 0);
#endif
    }

    // ---- Phase 0 (overlaps TDM): Wk/Wv -> bf16 A-fragment layout (ISA 16-bit A 16x32) ----
    for (int i = tid; i < 2 * C_DIM * C_DIM; i += 256) {
        int mat = i >> 12;
        int rem = i & 4095;
        int o = rem >> 6, c = rem & 63;
        float w = (mat ? Wv : Wk)[o * 64 + c];
        int mt = o >> 4, m = o & 15;
        int kstep = c >> 5, k = c & 31;
        int b15  = k & 15;
        int lane = m + ((b15 >> 3) << 4);               // K bit3 -> lane-half
        int j    = ((k >> 4) << 2) + ((b15 & 7) >> 1);  // VGPR index 0..7
        int e    = k & 1;
        Wf[(((mat * 4 + mt) * 2 + kstep) * 32 + lane) * 16 + 2 * j + e] = (__bf16)w;
    }

    // ---- Phase 3a (overlaps TDM): feat_q tile -> registers -> LDS ----
    {
        float tmp[4];
        #pragma unroll
        for (int t = 0; t < 4; ++t) {
            int i = tid + t * 256;
            tmp[t] = qbase[(long)(i >> 4) * qplane + (i & 15)];
        }
        #pragma unroll
        for (int t = 0; t < 4; ++t) {
            int i = tid + t * 256;
            fqs[(i >> 4) * QT + (i & 15)] = tmp[t];
        }
    }

    // TDM completion (TENSORcnt only nonzero on wave 0), then release all waves.
    __builtin_amdgcn_s_wait_tensorcnt(0);
    __syncthreads();

    // ---- Phase 1: staged fp32 tile -> bf16 B-fragment layout ----
    // Stage layout is tile-linear: ch*400 + row*80 + col (x fastest).
    // B layout (table order): K bit3 -> lane-half; element = (K&7) + 8*(K>>4).
    {
        const float4* Xs4 = (const float4*)Ks;
        for (int i = tid; i < C_DIM * PX / 4; i += 256) {
            int ch = i / (PX / 4);
            int px0 = (i - ch * (PX / 4)) * 4;
            const float4 g = Xs4[i];
            int kstep = ch >> 5, k = ch & 31;
            int lane_k = ((k >> 3) & 1) << 4;
            int j = ((k >> 4) << 2) + ((k & 7) >> 1);
            int e = k & 1;
            #pragma unroll
            for (int t = 0; t < 4; ++t) {
                int px = px0 + t;
                int nt = px >> 4, nlo = px & 15;
                float gv = (t == 0) ? g.x : (t == 1) ? g.y : (t == 2) ? g.z : g.w;
                Xf[((kstep * NTILE + nt) * 32 + (nlo + lane_k)) * 16 + 2 * j + e] = (__bf16)gv;
            }
        }
    }
    __syncthreads();

    // ---- Phase 2: K = Wk*X, V = Wv*X via v_wmma_f32_16x16x32_bf16 (overwrites stage) ----
    {
        const int wave = tid >> 5;
        const int lane = tid & 31;
        for (int job = wave; job < 2 * 4 * NTILE; job += 8) {   // 200 jobs, 25/wave
            int mat = job / (4 * NTILE);
            int rem = job - mat * 4 * NTILE;
            int mt  = rem / NTILE;
            int nt  = rem - mt * NTILE;
            const __bf16* wfb = Wf + (size_t)((mat * 4 + mt) * 2) * 512;
            const __bf16* xfb = Xf + (size_t)nt * 512;
            v16bf a0 = *(const v16bf*)(wfb + lane * 16);
            v16bf a1 = *(const v16bf*)(wfb + 512 + lane * 16);
            v16bf b0 = *(const v16bf*)(xfb + lane * 16);
            v16bf b1 = *(const v16bf*)(xfb + (size_t)NTILE * 512 + lane * 16);
            v8f acc = {};
            acc = __builtin_amdgcn_wmma_f32_16x16x32_bf16(false, a0, false, b0,
                                                          (short)0, acc, false, false);
            acc = __builtin_amdgcn_wmma_f32_16x16x32_bf16(false, a1, false, b1,
                                                          (short)0, acc, false, false);
            float* dst = mat ? Vs : Ks;
            int n   = lane & 15;
            int mhi = (lane >> 4) << 3;
            int px  = nt * 16 + n;
            #pragma unroll
            for (int r = 0; r < 8; ++r)
                dst[(mt * 16 + mhi + r) * PX + px] = acc[r];
        }
    }
    __syncthreads();   // Ks/Vs ready

    // ---- Phase 3b: q projection (tiny: 16px x 64ch), pre-scale by 1/sqrt(C)=1/8 ----
    for (int i = tid; i < C_DIM * QT; i += 256) {
        int o = i >> 4, l = i & 15;
        float acc = 0.f;
        #pragma unroll 8
        for (int c = 0; c < C_DIM; ++c)
            acc = fmaf(Wq[o * 64 + c], fqs[c * QT + l], acc);
        qs[l * C_DIM + o] = acc * 0.125f;
    }
    __syncthreads();

    // ---- Phase 4: attention logits: attn[l][n] = q[l] . K[:,px(l,n)] ----
    for (int i = tid; i < QT * 25; i += 256) {
        int l = i / 25, n = i - l * 25;
        int kh = n / 5, kw = n - kh * 5;
        int px = kh * COLS + l * KWIN + kw;
        const float* kc = Ks + px;
        const float* ql = qs + l * C_DIM;
        float acc = 0.f;
        #pragma unroll 8
        for (int c = 0; c < C_DIM; ++c)
            acc = fmaf(ql[c], kc[c * PX], acc);
        attnB[l * 25 + n] = acc;
    }
    __syncthreads();

    // ---- Phase 5: softmax over the 25-wide window ----
    if (tid < QT) {
        float* a = attnB + tid * 25;
        float mx = a[0];
        #pragma unroll
        for (int n = 1; n < 25; ++n) mx = fmaxf(mx, a[n]);
        float s = 0.f;
        #pragma unroll
        for (int n = 0; n < 25; ++n) { float e = __expf(a[n] - mx); a[n] = e; s += e; }
        float inv = 1.f / s;
        #pragma unroll
        for (int n = 0; n < 25; ++n) a[n] *= inv;
    }
    __syncthreads();

    // ---- Phase 6: out = attn @ V + residual ----
    for (int i = tid; i < C_DIM * QT; i += 256) {
        int c = i >> 4, l = i & 15;
        const float* vc = Vs + (long)c * PX;
        const float* a  = attnB + l * 25;
        float acc = 0.f;
        #pragma unroll
        for (int n = 0; n < 25; ++n) {
            int kh = n / 5, kw = n - kh * 5;
            acc = fmaf(a[n], vc[kh * COLS + l * KWIN + kw], acc);
        }
        obase[(long)c * qplane + l] = fqs[c * QT + l] + acc;
    }
}

extern "C" void kernel_launch(void* const* d_in, const int* in_sizes, int n_in,
                              void* d_out, int out_size, void* d_ws, size_t ws_size,
                              hipStream_t stream) {
    const float* feat_q  = (const float*)d_in[0];
    const float* feat_kv = (const float*)d_in[1];
    const float* Wq      = (const float*)d_in[2];
    const float* Wk      = (const float*)d_in[3];
    const float* Wv      = (const float*)d_in[4];
    float* out = (float*)d_out;

    const int B = in_sizes[0] / (C_DIM * H_DIM * W_DIM);

    size_t smem = (size_t)KS_OFF                         // Xf + Wf (bf16)
                + (size_t)2 * C_DIM * PX * sizeof(float) // Ks, Vs (Ks doubles as fp32 stage)
                + (size_t)QT * C_DIM * sizeof(float)     // qs
                + (size_t)C_DIM * QT * sizeof(float)     // fqs
                + (size_t)QT * 25 * sizeof(float);       // attnB   => 282176 B

    dim3 grid(W_DIM / QT, H_DIM, B);
    cag_local_fused<<<grid, 256, smem, stream>>>(feat_q, feat_kv, Wq, Wk, Wv, out);
}
